// GNNI_23948737643236
// MI455X (gfx1250) — compile-verified, matrix operand order checked by reading, not run on hardware
//
#include <hip/hip_runtime.h>
#include <hip/hip_bf16.h>

typedef __attribute__((ext_vector_type(16))) _Float16 v16h;
typedef __attribute__((ext_vector_type(8)))  float    v8f;

#define GROWS 256
#define GCOLS 512
#define GBATCH 512
#define GDEG 16
#define NPG (GROWS + GCOLS)   /* 768 nodes per graph  */
#define EPG (GROWS * GDEG)    /* 4096 edges per graph */
#define NPASS 10              /* NC=5 iterations x 2 directions */

__device__ __forceinline__ float sigmoidf_fast(float x) {
    return 1.0f / (1.0f + __expf(-x));
}

__global__ __launch_bounds__(256, 1) void ldpc_gnn_kernel(
    const float* __restrict__ x,
    const int*   __restrict__ eidx,   // [2, E] (row0=src, row1=dst-before-shift)
    const float* __restrict__ w1_1, const float* __restrict__ b1_1,
    const float* __restrict__ w2_1, const float* __restrict__ b2_1,
    const float* __restrict__ wih_1, const float* __restrict__ whh_1,
    const float* __restrict__ bih_1, const float* __restrict__ bhh_1,
    const float* __restrict__ w1_2, const float* __restrict__ b1_2,
    const float* __restrict__ w2_2, const float* __restrict__ b2_2,
    const float* __restrict__ wih_2, const float* __restrict__ whh_2,
    const float* __restrict__ bih_2, const float* __restrict__ bhh_2,
    float* __restrict__ out)
{
    __shared__ float    sh[NPG];          // node state h (resident all 10 rounds)
    __shared__ float    sm[NPG];          // aggregated messages m
    __shared__ unsigned sedge[EPG];       // packed (src_local | (dst_local+ROWS)<<16)

    const int b    = blockIdx.x;
    const int t    = threadIdx.x;
    const int lane = t & 31;
    const int wv   = t >> 5;
    const int nodeBase = b * NPG;
    const int edgeBase = b * EPG;
    const int Etot     = EPG * GBATCH;

    // ---- Phase 0: stage graph into LDS (global memory touched once) ----
    for (int n = t; n < NPG; n += 256) sh[n] = x[nodeBase + n];
    for (int e = t; e < EPG; e += 256) {
        int s = eidx[edgeBase + e] - nodeBase;                // check id, [0,256)
        int d = eidx[Etot + edgeBase + e] - nodeBase + GROWS; // var id,  [256,768)
        sedge[e] = (unsigned)(s | (d << 16));
    }

    // ---- Static WMMA A operand: augmented layer-1 weights, per direction. ----
    // D = A(16x32) x B(32x16):  A row M = hidden unit, K = [W1[m,0], W1[m,1], b1[m], 0...]
    // 16-bit A layout: lanes 0-15 hold M=lane, VGPR0 packs K0,K1; VGPR1 packs K2,K3.
    // Lanes 16-31 hold K=8..15 -> all zero here.
    const int  msel = lane & 15;
    const bool al   = (lane < 16) && (msel < 10);
    v16h A0 = {};
    v16h A1 = {};
    if (al) {
        A0[0] = (_Float16)w1_1[msel * 2 + 0];
        A0[1] = (_Float16)w1_1[msel * 2 + 1];
        A0[2] = (_Float16)b1_1[msel];
        A1[0] = (_Float16)w1_2[msel * 2 + 0];
        A1[1] = (_Float16)w1_2[msel * 2 + 1];
        A1[2] = (_Float16)b1_2[msel];
    }

    // ---- Per-lane layer-2 weights: lane half selects unit range 0-7 / 8-15 ----
    const int uoff = (lane >> 4) << 3;     // 0 for lanes<16, 8 for lanes>=16
    float w2c0[8], w2c1[8];
#pragma unroll
    for (int r = 0; r < 8; ++r) {
        int u = r + uoff;
        w2c0[r] = (u < 10) ? w2_1[u] : 0.0f;
        w2c1[r] = (u < 10) ? w2_2[u] : 0.0f;
    }
    const float b2a = b2_1[0];
    const float b2b = b2_2[0];

    // GRU weights (uniform scalars, both directions)
    float WI0[3], WH0[3], BI0[3], BH0[3], WI1[3], WH1[3], BI1[3], BH1[3];
#pragma unroll
    for (int g = 0; g < 3; ++g) {
        WI0[g] = wih_1[g]; WH0[g] = whh_1[g]; BI0[g] = bih_1[g]; BH0[g] = bhh_1[g];
        WI1[g] = wih_2[g]; WH1[g] = whh_2[g]; BI1[g] = bih_2[g]; BH1[g] = bhh_2[g];
    }

    __syncthreads();

    // ---- 10 message-passing rounds entirely in LDS ----
    for (int it = 0; it < NPASS; ++it) {
        const int  dir = it & 1;          // 0: ggc1 (update vars), 1: ggc2 (update checks)
        const v16h Aw  = dir ? A1 : A0;
        const float b2v = dir ? b2b : b2a;
        float w2w[8];
#pragma unroll
        for (int r = 0; r < 8; ++r) w2w[r] = dir ? w2c1[r] : w2c0[r];

        for (int n = t; n < NPG; n += 256) sm[n] = 0.0f;
        __syncthreads();

        // Edge loop: 16 edges per wave-tile, 32 uniform tiles per wave.
        for (int eb = wv * 16; eb < EPG; eb += 128) {
            // B matrix (32x16 f16): lane n<16 = edge column eb+n: K0=h_j, K1=h_i, K2=1
            v16h bm = {};
            unsigned p = 0;
            if (lane < 16) {
                p = sedge[eb + lane];
                int s = (int)(p & 0xffffu);
                int d = (int)(p >> 16);
                int j = dir ? d : s;
                int i = dir ? s : d;
                bm[0] = (_Float16)sh[j];
                bm[1] = (_Float16)sh[i];
                bm[2] = (_Float16)1.0f;
            }
            v8f c = {};
            c = __builtin_amdgcn_wmma_f32_16x16x32_f16(
                    false, Aw, false, bm, (short)0, c, false, false);

            // Layer 2 in registers: this lane holds hidden units uoff..uoff+7 of
            // its edge column; relu + dot with w2, then join the two unit halves.
            float partial = 0.0f;
#pragma unroll
            for (int r = 0; r < 8; ++r)
                partial += fmaxf(c[r], 0.0f) * w2w[r];
            partial += __shfl_xor(partial, 16, 32);

            // Scatter: owning lane already has the packed edge word.
            if (lane < 16) {
                int agg = dir ? (int)(p & 0xffffu) : (int)(p >> 16);
                atomicAdd(&sm[agg], partial + b2v);   // ds_add_f32
            }
        }
        __syncthreads();

        // GRU update at aggregated side (input = old h, hidden = message sum)
        const int cnt   = dir ? GROWS : GCOLS;
        const int base0 = dir ? 0 : GROWS;
        float wi0 = dir ? WI1[0] : WI0[0], wh0 = dir ? WH1[0] : WH0[0];
        float bi0 = dir ? BI1[0] : BI0[0], bh0 = dir ? BH1[0] : BH0[0];
        float wi1 = dir ? WI1[1] : WI0[1], wh1 = dir ? WH1[1] : WH0[1];
        float bi1 = dir ? BI1[1] : BI0[1], bh1 = dir ? BH1[1] : BH0[1];
        float wi2 = dir ? WI1[2] : WI0[2], wh2 = dir ? WH1[2] : WH0[2];
        float bi2 = dir ? BI1[2] : BI0[2], bh2 = dir ? BH1[2] : BH0[2];
        for (int k = t; k < cnt; k += 256) {
            int   nidx = base0 + k;
            float xg = sh[nidx];
            float hx = sm[nidx];
            float r  = sigmoidf_fast(xg * wi0 + bi0 + hx * wh0 + bh0);
            float z  = sigmoidf_fast(xg * wi1 + bi1 + hx * wh1 + bh1);
            float nn = tanhf(xg * wi2 + bi2 + r * (hx * wh2 + bh2));
            sh[nidx] = (1.0f - z) * nn + z * hx;
        }
        __syncthreads();
    }

    // ---- Output: clip(sigmoid(-h), eps, 1-eps), one coalesced store pass ----
    for (int n = t; n < NPG; n += 256) {
        float v = sigmoidf_fast(-sh[n]);
        v = fminf(fmaxf(v, 1e-7f), 1.0f - 1e-7f);
        out[nodeBase + n] = v;
    }
}

extern "C" void kernel_launch(void* const* d_in, const int* in_sizes, int n_in,
                              void* d_out, int out_size, void* d_ws, size_t ws_size,
                              hipStream_t stream) {
    (void)in_sizes; (void)n_in; (void)out_size; (void)d_ws; (void)ws_size;
    const float* x    = (const float*)d_in[0];
    const int*   eidx = (const int*)d_in[1];
    ldpc_gnn_kernel<<<GBATCH, 256, 0, stream>>>(
        x, eidx,
        (const float*)d_in[2],  (const float*)d_in[3],
        (const float*)d_in[4],  (const float*)d_in[5],
        (const float*)d_in[6],  (const float*)d_in[7],
        (const float*)d_in[8],  (const float*)d_in[9],
        (const float*)d_in[10], (const float*)d_in[11],
        (const float*)d_in[12], (const float*)d_in[13],
        (const float*)d_in[14], (const float*)d_in[15],
        (const float*)d_in[16], (const float*)d_in[17],
        (float*)d_out);
}
